// Attention_37769942401553
// MI455X (gfx1250) — matrix operand order repaired and checked by reading, not compiled, not run on hardware
//
#include <hip/hip_runtime.h>
#include <math.h>

typedef __attribute__((ext_vector_type(16))) __bf16       v16bf;
typedef __attribute__((ext_vector_type(2)))  __bf16       v2bf;
typedef __attribute__((ext_vector_type(8)))  float        v8f;
typedef __attribute__((ext_vector_type(8)))  unsigned int v8u;

#define BB 32
#define SS 1024
#define HH 512
#define KK 1024   /* 2H */

// ---- packed f32x2 -> bf16x2 conversion (v_cvt_pk_bf16_f32 on gfx1250) ----
static __device__ __forceinline__ v2bf pkbf(float a, float b) {
#if __has_builtin(__builtin_amdgcn_cvt_pk_bf16_f32)
  return __builtin_amdgcn_cvt_pk_bf16_f32(a, b);
#else
  v2bf r; r[0] = (__bf16)a; r[1] = (__bf16)b; return r;
#endif
}

static __device__ __forceinline__ float fast_tanh(float x) {
#if __has_builtin(__builtin_amdgcn_tanhf)
  return __builtin_amdgcn_tanhf(x);
#elif __has_builtin(__builtin_amdgcn_tanh_f32)
  return __builtin_amdgcn_tanh_f32(x);
#else
  return tanhf(x);
#endif
}

// ---------------- Kernel 1: hpb[b,h] = bias[h] + sum_k hidden[b,k]*Wh[k,h] ----------------
__global__ void __launch_bounds__(256)
hpb_kernel(const float* __restrict__ hidden, const float* __restrict__ W,
           const float* __restrict__ bias, float* __restrict__ hpb) {
  int idx = blockIdx.x * 256 + threadIdx.x;   // 0..16383
  int b = idx >> 9;
  int h = idx & 511;
  const float* hrow = hidden + b * HH;
  float acc = bias[h];
#pragma unroll 4
  for (int k = 0; k < HH; ++k)
    acc = fmaf(hrow[k], W[k * HH + h], acc);
  hpb[idx] = acc;
}

// ---------------- Kernel 2: pack We (f32, row-major KxN) into bf16 WMMA B fragments ------
// Fragment (nt, ks): N = nt*16..+15, K = ks*32..+31.
// Lane l<16: N=l, K=0..15 (2/dword); lane l>=16: N=l-16, K=16..31.
// Lane dwords contiguous at frag*256 + lane*8.
__global__ void __launch_bounds__(256)
pack_kernel(const float* __restrict__ W, unsigned int* __restrict__ Bp) {
  int idx  = blockIdx.x * 256 + threadIdx.x;  // 32768 threads
  int lane = idx & 31;
  int frag = idx >> 5;                        // nt*32 + ks
  int ks   = frag & 31;
  int nt   = frag >> 5;
  int n    = nt * 16 + (lane & 15);
  int kb   = ks * 32 + ((lane < 16) ? 0 : 16);
  const float* We = W + HH * HH;              // skip Wh rows
  unsigned int* dst = Bp + (size_t)frag * 256 + lane * 8;
#pragma unroll
  for (int v = 0; v < 8; ++v) {
    float f0 = We[(size_t)(kb + 2 * v)     * HH + n];
    float f1 = We[(size_t)(kb + 2 * v + 1) * HH + n];
    dst[v] = __builtin_bit_cast(unsigned int, pkbf(f0, f1));
  }
}

// ---------------- Kernel 3: fused e_proj GEMM + tanh + dot(v) -> scores ------------------
// WG = 256 threads (8 waves) covers 32 S-rows x 512 H cols.
// Wave w: M-tiles {0,1} (all 32 rows), N-tiles {4w..4w+3} (cols 64w..64w+63).
// A (32 x 64 f32 chunk) staged global->LDS with async b128 copies, double buffered.
// B fragments software-pipelined one K-step ahead (ping-pong on kh parity).
#define AROW 68   /* 64 dwords + 4 pad: lane banks = 4*lane + c -> conflict free */

__global__ void __launch_bounds__(256)
attn_kernel(const float* __restrict__ enc, const float* __restrict__ hpb,
            const unsigned int* __restrict__ Bp, const float* __restrict__ v,
            float* __restrict__ scores) {
  __shared__ float sA[2][32 * AROW];
  __shared__ float sscore[32];

  const int tid   = threadIdx.x;
  const int wave  = tid >> 5;
  const int lane  = tid & 31;
  const int s0    = blockIdx.x * 32;   // S tile (32 rows)
  const int batch = blockIdx.y;

  if (tid < 32) sscore[tid] = 0.0f;

  // --- async staging: thread t copies 32B (2 x b128) of the 32x64 f32 chunk ---
  const int srow = tid >> 3;          // 0..31
  const int sseg = tid & 7;           // 32B segment within the 64-dword row
  const int gbase = 4 * ((batch * SS + s0 + srow) * KK + sseg * 8);
  const unsigned lb0 = (unsigned)(size_t)(&sA[0][srow * AROW + sseg * 8]);
  const unsigned lb1 = (unsigned)(size_t)(&sA[1][srow * AROW + sseg * 8]);

  // prologue: stage chunk 0 into buffer 0
  asm volatile("global_load_async_to_lds_b128 %0, %1, %2"
               :: "v"(lb0), "v"(gbase), "s"(enc) : "memory");
  asm volatile("global_load_async_to_lds_b128 %0, %1, %2"
               :: "v"(lb0 + 16u), "v"(gbase + 16), "s"(enc) : "memory");

  // --- A fragment addressing within LDS chunk (bf16 16x32 A layout) ---
  const int m   = lane & 15;
  const int kbA = (lane < 16) ? 0 : 16;

  // B fragment loader: 2 x b128 per tile, lane dwords contiguous
  const unsigned int* blane = Bp + (size_t)lane * 8;
#define LOADB(ksv, dst)                                                        \
  {                                                                            \
    _Pragma("unroll")                                                          \
    for (int t = 0; t < 4; ++t)                                                \
      (dst)[t] = *reinterpret_cast<const v8u*>(                                \
          blane + ((size_t)((wave * 4 + t) * 32 + (ksv))) * 256);              \
  }

  v8f c[2][4];
#pragma unroll
  for (int mt = 0; mt < 2; ++mt)
#pragma unroll
    for (int t = 0; t < 4; ++t) c[mt][t] = (v8f){};

  v8u bb[2][4];
  LOADB(0, bb[0]);   // prefetch B for ks = 0

  for (int chunk = 0; chunk < 16; ++chunk) {
    const int buf = chunk & 1;
    if (chunk + 1 < 16) {
      // stage next chunk into the other buffer (fully read one iteration ago)
      int goff = gbase + (chunk + 1) * 64 * 4;
      unsigned lb = buf ? lb0 : lb1;
      asm volatile("global_load_async_to_lds_b128 %0, %1, %2"
                   :: "v"(lb), "v"(goff), "s"(enc) : "memory");
      asm volatile("global_load_async_to_lds_b128 %0, %1, %2"
                   :: "v"(lb + 16u), "v"(goff + 16), "s"(enc) : "memory");
      asm volatile("s_wait_asynccnt 0x2" ::: "memory");
    } else {
      asm volatile("s_wait_asynccnt 0x0" ::: "memory");
    }
    __syncthreads();   // current buffer visible to all waves

#pragma unroll
    for (int kh = 0; kh < 2; ++kh) {            // two K=32 steps per chunk
      const int ks = chunk * 2 + kh;            // global K-step (ks & 1 == kh)

      // A fragments for both M-tiles from LDS (+ packed cvt to bf16)
      v16bf a[2];
#pragma unroll
      for (int mt = 0; mt < 2; ++mt) {
        const float4* pA = reinterpret_cast<const float4*>(
            &sA[buf][(mt * 16 + m) * AROW + kh * 32 + kbA]);
        float4 f0 = pA[0];   // K kb+0..3
        float4 f1 = pA[1];   // K kb+4..7
        float4 f2 = pA[2];   // K kb+8..11
        float4 f3 = pA[3];   // K kb+12..15
        union { v16bf v; v2bf p[8]; } ua;
        ua.p[0] = pkbf(f0.x, f0.y); ua.p[1] = pkbf(f0.z, f0.w);
        ua.p[2] = pkbf(f1.x, f1.y); ua.p[3] = pkbf(f1.z, f1.w);
        ua.p[4] = pkbf(f2.x, f2.y); ua.p[5] = pkbf(f2.z, f2.w);
        ua.p[6] = pkbf(f3.x, f3.y); ua.p[7] = pkbf(f3.z, f3.w);
        a[mt] = ua.v;
      }

      // prefetch B for the next K-step while this step's WMMAs run
      if (ks + 1 < 32) LOADB(ks + 1, bb[1 - kh]);

#pragma unroll
      for (int t = 0; t < 4; ++t) {
        v16bf bm = __builtin_bit_cast(v16bf, bb[kh][t]);
#pragma unroll
        for (int mt = 0; mt < 2; ++mt)
          c[mt][t] = __builtin_amdgcn_wmma_f32_16x16x32_bf16(
              false, a[mt], false, bm, (short)0, c[mt][t], false, false);
      }
    }
    __syncthreads();   // everyone done reading this buffer before restage
  }
#undef LOADB

  // Epilogue: C layout — lane gives N=(lane&15), VGPR r gives M=r+(lane>=16?8:0)
  float racc[2][8];
#pragma unroll
  for (int mt = 0; mt < 2; ++mt)
#pragma unroll
    for (int r = 0; r < 8; ++r) racc[mt][r] = 0.0f;

  const int ncol = lane & 15;
#pragma unroll
  for (int t = 0; t < 4; ++t) {
    int col  = wave * 64 + t * 16 + ncol;
    float hb = hpb[batch * HH + col];
    float vv = v[col];
#pragma unroll
    for (int mt = 0; mt < 2; ++mt)
#pragma unroll
      for (int r = 0; r < 8; ++r)
        racc[mt][r] += fast_tanh(c[mt][t][r] + hb) * vv;
  }

  // Reduce over the 16 lanes of each half-wave (same M set within a half)
#pragma unroll
  for (int mask = 1; mask < 16; mask <<= 1) {
#pragma unroll
    for (int mt = 0; mt < 2; ++mt)
#pragma unroll
      for (int r = 0; r < 8; ++r)
        racc[mt][r] += __shfl_xor(racc[mt][r], mask, 32);
  }

  if (lane == 0 || lane == 16) {
    int base = (lane == 0) ? 0 : 8;
#pragma unroll
    for (int mt = 0; mt < 2; ++mt)
#pragma unroll
      for (int r = 0; r < 8; ++r)
        atomicAdd(&sscore[mt * 16 + base + r], racc[mt][r]);
  }
  __syncthreads();
  if (tid < 32)
    scores[batch * SS + s0 + tid] = sscore[tid];
}

// ---------------- Kernel 4: softmax over S per batch -------------------------------------
__global__ void __launch_bounds__(256)
softmax_kernel(const float* __restrict__ scores, float* __restrict__ out) {
  __shared__ float red[8];
  const int b    = blockIdx.x;
  const int tid  = threadIdx.x;
  const int wave = tid >> 5;
  const int lane = tid & 31;

  float x[4];
#pragma unroll
  for (int j = 0; j < 4; ++j) x[j] = scores[b * SS + tid + 256 * j];

  float mx = fmaxf(fmaxf(x[0], x[1]), fmaxf(x[2], x[3]));
#pragma unroll
  for (int mask = 1; mask < 32; mask <<= 1)
    mx = fmaxf(mx, __shfl_xor(mx, mask, 32));
  if (lane == 0) red[wave] = mx;
  __syncthreads();
  float m8 = red[0];
#pragma unroll
  for (int w = 1; w < 8; ++w) m8 = fmaxf(m8, red[w]);
  __syncthreads();

  float e[4];
  float s = 0.0f;
#pragma unroll
  for (int j = 0; j < 4; ++j) { e[j] = __expf(x[j] - m8); s += e[j]; }
#pragma unroll
  for (int mask = 1; mask < 32; mask <<= 1)
    s += __shfl_xor(s, mask, 32);
  if (lane == 0) red[wave] = s;
  __syncthreads();
  float st = 0.0f;
#pragma unroll
  for (int w = 0; w < 8; ++w) st += red[w];
  float inv = 1.0f / st;
#pragma unroll
  for (int j = 0; j < 4; ++j) out[b * SS + tid + 256 * j] = e[j] * inv;
}

extern "C" void kernel_launch(void* const* d_in, const int* in_sizes, int n_in,
                              void* d_out, int out_size, void* d_ws, size_t ws_size,
                              hipStream_t stream) {
  const float* hidden = (const float*)d_in[0];   // (32, 512)
  const float* enc    = (const float*)d_in[1];   // (32, 1024, 1024)
  const float* W      = (const float*)d_in[2];   // (1536, 512)
  const float* bias   = (const float*)d_in[3];   // (512,)
  const float* v      = (const float*)d_in[4];   // (512,)
  float* out = (float*)d_out;                    // (32, 1024)

  char* ws = (char*)d_ws;
  float*        hpb    = (float*)ws;                          // 64 KB
  unsigned int* Bp     = (unsigned int*)(ws + 65536);         // 1 MB bf16-packed We
  float*        scores = (float*)(ws + 65536 + 1048576);      // 128 KB

  hpb_kernel    <<<64,            256, 0, stream>>>(hidden, W, bias, hpb);
  pack_kernel   <<<128,           256, 0, stream>>>(W, Bp);
  attn_kernel   <<<dim3(32, 32),  256, 0, stream>>>(enc, hpb, Bp, v, scores);
  softmax_kernel<<<32,            256, 0, stream>>>(scores, out);
}
